// SparseEncoder_63161789055543
// MI455X (gfx1250) — matrix operand order repaired and checked by reading, not compiled, not run on hardware
//
#include <hip/hip_runtime.h>
#include <hip/hip_bf16.h>

// ---------------- problem constants (match setup_inputs) ----------------
#define TOK 4096      // batch*seq = 2*2048
#define DM  1024      // d_model
#define DC  16384     // d_concepts
#define KSEL 32       // top-k

typedef __attribute__((ext_vector_type(16))) __bf16 v16bf;
typedef __attribute__((ext_vector_type(8)))  float  v8f;

union FragU { uint4 u[2]; v16bf v; };

// ---------------- fp32 -> bf16 (round to nearest even) ----------------
__device__ __forceinline__ unsigned short f2bf(float f) {
    unsigned int u = __float_as_uint(f);
    unsigned int r = u + 0x7FFFu + ((u >> 16) & 1u);
    return (unsigned short)(r >> 16);
}

__global__ void convert_f32_bf16(const float* __restrict__ in,
                                 unsigned short* __restrict__ out, int n) {
    int i = blockIdx.x * blockDim.x + threadIdx.x;
    if (i < n) out[i] = f2bf(in[i]);
}

// ---------------- W_emb [DM x DC] -> W_embT [DC x DM] (tiled) ----------------
__global__ __launch_bounds__(256)
void transpose_wemb(const float* __restrict__ in, float* __restrict__ outT) {
    __shared__ float tile[32][33];
    int c = blockIdx.x * 32 + threadIdx.x;   // concept col in source
    #pragma unroll
    for (int i = 0; i < 32; i += 8) {
        int d = blockIdx.y * 32 + threadIdx.y + i;
        tile[threadIdx.y + i][threadIdx.x] = in[(size_t)d * DC + c];
    }
    __syncthreads();
    int d = blockIdx.y * 32 + threadIdx.x;   // d_model col in dest
    #pragma unroll
    for (int i = 0; i < 32; i += 8) {
        int cc = blockIdx.x * 32 + threadIdx.y + i;
        outT[(size_t)cc * DM + d] = tile[threadIdx.x][threadIdx.y + i];
    }
}

// ---------------- encode GEMM: pre[t,c] = sum_d A[t,d]*W[c,d] + bias[c] ----
// block = 256 threads = 8 waves (2 x 4 wave grid); block tile 128M x 128N,
// wave tile 64M x 32N = 4x2 subtiles of 16x16, K-step 32 (bf16 WMMA).
__global__ __launch_bounds__(256)
void encode_wmma(const unsigned short* __restrict__ A,
                 const unsigned short* __restrict__ W,
                 const float* __restrict__ bias,
                 float* __restrict__ pre) {
    const int lane = threadIdx.x & 31;
    const int wv   = threadIdx.x >> 5;
    const int wm   = wv >> 2;            // 0..1
    const int wn   = wv & 3;             // 0..3
    const int mBase = blockIdx.y * 128 + wm * 64;
    const int nBase = blockIdx.x * 128 + wn * 32;
    const int lr = lane & 15;
    const int hi = lane >> 4;

    // A fragment rows: lane holds row (mBase+mi*16+lr), K runs 0..15 contiguous
    // shifted by +16 for hi lanes (matches B layout).
    const unsigned short* aRow[4];
    #pragma unroll
    for (int mi = 0; mi < 4; ++mi)
        aRow[mi] = A + (size_t)(mBase + mi * 16 + lr) * DM + hi * 16;

    // B fragment: lane holds 16 contiguous K of W row (nBase+ni*16+lr),
    // hi lanes offset by +16 in K (same contiguous K order as A).
    const unsigned short* bRow[2];
    #pragma unroll
    for (int ni = 0; ni < 2; ++ni)
        bRow[ni] = W + (size_t)(nBase + ni * 16 + lr) * DM + hi * 16;

    v8f acc[4][2] = {};

    for (int k0 = 0; k0 < DM; k0 += 32) {
        FragU a[4], b[2];
        #pragma unroll
        for (int mi = 0; mi < 4; ++mi) {
            a[mi].u[0] = *(const uint4*)(aRow[mi] + k0);        // K +0..7
            a[mi].u[1] = *(const uint4*)(aRow[mi] + k0 + 8);    // K +8..15
        }
        #pragma unroll
        for (int ni = 0; ni < 2; ++ni) {
            b[ni].u[0] = *(const uint4*)(bRow[ni] + k0);        // K +0..7
            b[ni].u[1] = *(const uint4*)(bRow[ni] + k0 + 8);    // K +8..15
        }
        #pragma unroll
        for (int mi = 0; mi < 4; ++mi)
            #pragma unroll
            for (int ni = 0; ni < 2; ++ni)
                acc[mi][ni] = __builtin_amdgcn_wmma_f32_16x16x32_bf16(
                    false, a[mi].v, false, b[ni].v,
                    (short)0, acc[mi][ni], false, false);
    }

    // bias + store pre_act (C/D layout: lane holds col n=lr; VGPR r is row
    // mBase+mi*16+r (+8 for hi lanes)).
    #pragma unroll
    for (int ni = 0; ni < 2; ++ni) {
        const int c = nBase + ni * 16 + lr;
        const float bv = bias[c];
        #pragma unroll
        for (int mi = 0; mi < 4; ++mi) {
            #pragma unroll
            for (int r = 0; r < 8; ++r) {
                const int m = mBase + mi * 16 + r + hi * 8;
                pre[(size_t)m * DC + c] = acc[mi][ni][r] + bv;
            }
        }
    }
}

// ---------------- per-token top-32 + sparse decode ----------------
// One 256-thread workgroup per token. Each thread owns 64 pre_act values
// in registers; iterative global argmax extraction, then decode gather.
__global__ __launch_bounds__(256)
void topk_decode(const float* __restrict__ pre,
                 const float* __restrict__ wembT,
                 float* __restrict__ out) {
    const int t    = blockIdx.x;
    const int tid  = threadIdx.x;
    const int lane = tid & 31;
    const int wv   = tid >> 5;
    const float* row = pre + (size_t)t * DC;

    float v[64];
    #pragma unroll
    for (int j = 0; j < 64; ++j) v[j] = row[tid + 256 * j];

    unsigned long long used = 0ull;
    float bval = -3.402823466e38f;
    int   bj   = 0;
    #pragma unroll
    for (int j = 0; j < 64; ++j)
        if (v[j] > bval) { bval = v[j]; bj = j; }

    __shared__ float s_val[8];
    __shared__ int   s_idx[8];
    __shared__ float top_val[KSEL];
    __shared__ int   top_idx[KSEL];

    for (int iter = 0; iter < KSEL; ++iter) {
        // wave32 argmax reduce
        float rv = bval;
        int   ri = tid + 256 * bj;        // == concept index
        #pragma unroll
        for (int off = 16; off >= 1; off >>= 1) {
            float ov = __shfl_xor(rv, off, 32);
            int   oi = __shfl_xor(ri, off, 32);
            if (ov > rv) { rv = ov; ri = oi; }
        }
        if (lane == 0) { s_val[wv] = rv; s_idx[wv] = ri; }
        __syncthreads();
        // cross-wave argmax (redundant on all threads)
        float gv = s_val[0]; int gi = s_idx[0];
        #pragma unroll
        for (int w2 = 1; w2 < 8; ++w2)
            if (s_val[w2] > gv) { gv = s_val[w2]; gi = s_idx[w2]; }
        if (tid == 0) { top_val[iter] = gv; top_idx[iter] = gi; }
        // owning thread retires the winner and rescans its registers
        if (tid == (gi & 255)) {
            used |= 1ull << (gi >> 8);
            bval = -3.402823466e38f; bj = 0;
            #pragma unroll
            for (int j = 0; j < 64; ++j)
                if (!((used >> j) & 1ull) && v[j] > bval) { bval = v[j]; bj = j; }
        }
        __syncthreads();
    }

    // decode: out[t,d] = sum_j top_val[j] * W_embT[top_idx[j], d]
    float acc[4] = {0.f, 0.f, 0.f, 0.f};
    for (int j = 0; j < KSEL; ++j) {
        const float val = top_val[j];
        const float* wr = wembT + (size_t)top_idx[j] * DM;
        #pragma unroll
        for (int d4 = 0; d4 < 4; ++d4)
            acc[d4] += val * wr[tid + 256 * d4];
    }
    #pragma unroll
    for (int d4 = 0; d4 < 4; ++d4)
        out[(size_t)t * DM + tid + 256 * d4] = acc[d4];
}

// ---------------- launch ----------------
extern "C" void kernel_launch(void* const* d_in, const int* in_sizes, int n_in,
                              void* d_out, int out_size, void* d_ws, size_t ws_size,
                              hipStream_t stream) {
    const float* act   = (const float*)d_in[0];   // [TOK, DM]
    const float* wenc  = (const float*)d_in[1];   // [DC, DM]
    const float* wbias = (const float*)d_in[2];   // [DC]
    const float* wemb  = (const float*)d_in[3];   // [DM, DC]
    // d_in[4] = k (always 32 per setup_inputs)
    float* out = (float*)d_out;                   // [TOK, DM]

    char* ws = (char*)d_ws;
    unsigned short* actB  = (unsigned short*)(ws);                          //  8 MB
    unsigned short* wencB = (unsigned short*)(ws + (size_t)8  * (1 << 20)); // 32 MB
    float*          preact= (float*)         (ws + (size_t)40 * (1 << 20)); // 256 MB
    float*          wembT = (float*)         (ws + (size_t)296* (1 << 20)); // 64 MB

    // 1) fp32 -> bf16 conversions
    {
        int nA = TOK * DM;
        convert_f32_bf16<<<(nA + 255) / 256, 256, 0, stream>>>(act, actB, nA);
        int nW = DC * DM;
        convert_f32_bf16<<<(nW + 255) / 256, 256, 0, stream>>>(wenc, wencB, nW);
    }
    // 2) transpose W_emb for contiguous decode gathers
    {
        dim3 blk(32, 8), grd(DC / 32, DM / 32);
        transpose_wemb<<<grd, blk, 0, stream>>>(wemb, wembT);
    }
    // 3) encode GEMM (bf16 WMMA, f32 accumulate, fused bias)
    {
        dim3 grd(DC / 128, TOK / 128);
        encode_wmma<<<grd, 256, 0, stream>>>(actB, wencB, wbias, preact);
    }
    // 4) top-32 selection + sparse decode
    topk_decode<<<TOK, 256, 0, stream>>>(preact, wembT, out);
}
